// GraphTransformerLayer_60979945669316
// MI455X (gfx1250) — matrix operand order repaired and checked by reading, not compiled, not run on hardware
//
#include <hip/hip_runtime.h>
#include <hip/hip_bf16.h>
#include <math.h>

// Problem constants (match the reference)
#define NN_NODES 2048
#define NE 32768
#define DIM 256
#define NH 8
#define HDIM 32
#define DFF 1024

typedef __attribute__((ext_vector_type(16))) __bf16 v16bf;
typedef __attribute__((ext_vector_type(8)))  __bf16 v8bf;
typedef __attribute__((ext_vector_type(8)))  float  v8f;

__device__ __forceinline__ unsigned short bf_bits(float f) {
    // round-to-nearest-even fp32 -> bf16 bit pattern
    unsigned int u = __builtin_bit_cast(unsigned int, f);
    u += 0x7FFFu + ((u >> 16) & 1u);
    return (unsigned short)(u >> 16);
}

// ---------------------------------------------------------------------------
// elementwise fp32 -> bf16
// ---------------------------------------------------------------------------
__global__ void f32_to_bf16_kernel(const float* __restrict__ in,
                                   unsigned short* __restrict__ out, int n)
{
    int i = blockIdx.x * 256 + threadIdx.x;
    if (i < n) out[i] = bf_bits(in[i]);
}

// ---------------------------------------------------------------------------
// Pack fp32 weight B[K,Ncol] (row-major) into bf16 WMMA B-fragment order:
//   Bp[((panel * (K/32)) + ktile) * 512 + lane*16 + e]
// per CDNA5 ISA 7.12.2 (16-bit B 32x16): lane&15 = column within panel,
// e<8: K = 8*(lane>>4) + e; e>=8: K = 16 + 8*(lane>>4) + (e-8).
// ---------------------------------------------------------------------------
__global__ void pack_b_kernel(const float* __restrict__ B,
                              unsigned short* __restrict__ Bp,
                              int K, int Ncol)
{
    int d = blockIdx.x * 256 + threadIdx.x;
    int total = (Ncol >> 4) * (K >> 5) * 512;
    if (d >= total) return;
    int tile = d >> 9, r = d & 511;
    int lane = r >> 4, e = r & 15;
    int kt = tile % (K >> 5), p = tile / (K >> 5);
    int col = (p << 4) + (lane & 15);
    int k = (kt << 5) + ((e >> 3) << 4) + ((lane >> 4) << 3) + (e & 7);
    Bp[d] = bf_bits(B[(size_t)k * Ncol + col]);
}

// ---------------------------------------------------------------------------
// bf16 WMMA GEMM: C[M,Ncol] = act(A[M,K] @ B[K,Ncol] + bias)
// One wave32 per 16x64 tile (4 accumulators share each A fragment).
// Software-pipelined: fragments for step ks+1 are issued before the four
// WMMAs of step ks, so VMEM overlaps the XDL pipe (partial loadcnt waits).
// ---------------------------------------------------------------------------
template<int K>
__global__ void wmma_gemm_kernel(const unsigned short* __restrict__ A,
                                 const unsigned short* __restrict__ Bp,
                                 const float* __restrict__ bias,
                                 float* __restrict__ Cf,            // optional fp32 out
                                 unsigned short* __restrict__ Cb,   // optional bf16 out
                                 int M, int Ncol, int relu)
{
    constexpr int KT = K >> 5;
    const int wave   = blockIdx.x * (blockDim.x >> 5) + (threadIdx.x >> 5);
    const int lane   = threadIdx.x & 31;
    const int tilesN = Ncol >> 6;               // 64-wide N tiles
    const int tm = (wave / tilesN) << 4;
    const int tn = (wave % tilesN) << 6;
    if (tm >= M) return;

    const int half = lane >> 4;                 // 0 | 1
    const int l15  = lane & 15;
    const int ka   = half * 8;                  // A K-base within each 16-block

    const unsigned short* ap    = A + (size_t)(tm + l15) * K;
    const unsigned short* bbase = Bp + (size_t)(tn >> 4) * KT * 512 + lane * 16;

    auto load_a = [&](int ks) -> v16bf {
        const int k0 = ks << 5;
        v8bf lo = *reinterpret_cast<const v8bf*>(ap + k0 + ka);
        v8bf hi = *reinterpret_cast<const v8bf*>(ap + k0 + 16 + ka);
        return __builtin_shufflevector(lo, hi,
                   0,1,2,3,4,5,6,7,8,9,10,11,12,13,14,15);
    };
    auto load_b = [&](int t, int ks) -> v16bf {
        return *reinterpret_cast<const v16bf*>(bbase + (size_t)(t * KT + ks) * 512);
    };

    v8f acc[4] = {};
    // prologue: fragments for ks = 0
    v16bf a  = load_a(0);
    v16bf b0 = load_b(0, 0), b1 = load_b(1, 0), b2 = load_b(2, 0), b3 = load_b(3, 0);

#pragma unroll 2
    for (int ks = 0; ks < KT; ++ks) {
        v16bf an = a, bn0 = b0, bn1 = b1, bn2 = b2, bn3 = b3;
        if (ks + 1 < KT) {
            an  = load_a(ks + 1);
            bn0 = load_b(0, ks + 1);
            bn1 = load_b(1, ks + 1);
            bn2 = load_b(2, ks + 1);
            bn3 = load_b(3, ks + 1);
            __builtin_prefetch((const void*)(ap + ((ks + 1) << 5) + 32), 0, 3);
        }
        acc[0] = __builtin_amdgcn_wmma_f32_16x16x32_bf16(false, a, false, b0, (short)0, acc[0], false, false);
        acc[1] = __builtin_amdgcn_wmma_f32_16x16x32_bf16(false, a, false, b1, (short)0, acc[1], false, false);
        acc[2] = __builtin_amdgcn_wmma_f32_16x16x32_bf16(false, a, false, b2, (short)0, acc[2], false, false);
        acc[3] = __builtin_amdgcn_wmma_f32_16x16x32_bf16(false, a, false, b3, (short)0, acc[3], false, false);
        a = an; b0 = bn0; b1 = bn1; b2 = bn2; b3 = bn3;
    }

    const int baseRow = tm + half * 8;          // C/D: VGPR v -> row v (+8)
#pragma unroll
    for (int t = 0; t < 4; ++t) {
        const int col = tn + (t << 4) + l15;
        const float bv = bias[col];
#pragma unroll
        for (int v = 0; v < 8; ++v) {
            float o = acc[t][v] + bv;
            if (relu) o = fmaxf(o, 0.0f);
            const size_t off = (size_t)(baseRow + v) * Ncol + col;
            if (Cf) Cf[off] = o;
            if (Cb) Cb[off] = bf_bits(o);
        }
    }
}

// ---------------------------------------------------------------------------
// Graph structure.
//   eid[i*N+j] = max edge index with (src,dst)==(i,j), else -1
//                (last-write scatter semantics + dedup)
//   adj[i*N+j] = 1 iff connected (either direction or diagonal)
// ---------------------------------------------------------------------------
__global__ void init_graph_kernel(int* __restrict__ eid,
                                  unsigned char* __restrict__ adj)
{
    size_t i = (size_t)blockIdx.x * 256 + threadIdx.x;
    eid[i] = -1;
    int r = (int)(i >> 11);            // / 2048
    int c = (int)(i & (NN_NODES - 1));
    adj[i] = (r == c) ? 1 : 0;         // self-loops
}

__global__ void scatter_graph_kernel(const int* __restrict__ src,
                                     const int* __restrict__ dst,
                                     int* __restrict__ eid,
                                     unsigned char* __restrict__ adj)
{
    int e = blockIdx.x * 256 + threadIdx.x;
    if (e >= NE) return;
    const size_t s = (size_t)src[e], d = (size_t)dst[e];
    adj[s * NN_NODES + d] = 1;         // benign races: all write 1
    adj[d * NN_NODES + s] = 1;
    atomicMax(&eid[s * NN_NODES + d], e);
}

// edge_bias[E,H] = edge_feat[E,10] @ We[10,H] + be[H]  (tiny, VALU)
__global__ void edge_bias_kernel(const float* __restrict__ ef,
                                 const float* __restrict__ We,
                                 const float* __restrict__ be,
                                 float* __restrict__ out)
{
    int idx = blockIdx.x * 256 + threadIdx.x;
    if (idx >= NE * NH) return;
    int e = idx >> 3;      // H == 8
    int h = idx & 7;
    float s = be[h];
#pragma unroll
    for (int k = 0; k < 10; ++k) s += ef[e * 10 + k] * We[k * NH + h];
    out[idx] = s;
}

// ---------------------------------------------------------------------------
// Sparse masked attention, online softmax, bf16 output (feeds O-proj).
// One block (256 threads) per node i; wave w == head w, lane == head-dim.
// Candidate scan: lane d probes adj[i][jb+d] (coalesced 32B), __ballot
// compacts to a wave-uniform mask, wave iterates only over set bits.
// ---------------------------------------------------------------------------
__global__ void attn_kernel(const float* __restrict__ Q,
                            const float* __restrict__ Km,
                            const float* __restrict__ Vm,
                            const unsigned char* __restrict__ adj,
                            const int*  __restrict__ eid,
                            const float* __restrict__ ebias,
                            unsigned short* __restrict__ outb)
{
    const int i = blockIdx.x;
    const int h = threadIdx.x >> 5;   // head
    const int d = threadIdx.x & 31;   // dim in head
    const float scale = 0.17677669529663687f; // 1/sqrt(32)

    const float q = Q[(size_t)i * DIM + h * HDIM + d] * scale;
    float m = -INFINITY, l = 0.0f, acc = 0.0f;

    for (int jb = 0; jb < NN_NODES; jb += 32) {
        const unsigned char a = adj[(size_t)i * NN_NODES + jb + d];
        unsigned int mask = (unsigned int)__ballot(a != 0);   // wave32
        while (mask) {
            const int bit = __ffs(mask) - 1;
            mask &= mask - 1;
            const int j = jb + bit;

            float s = q * Km[(size_t)j * DIM + h * HDIM + d];
#pragma unroll
            for (int off = 16; off > 0; off >>= 1)
                s += __shfl_xor(s, off, 32);
            const int e_ij = eid[(size_t)i * NN_NODES + j];
            if (e_ij >= 0) s += ebias[(size_t)e_ij * NH + h];

            const float mn = fmaxf(m, s);
            const float c  = __expf(m - mn);   // 0 on first hit (m=-inf)
            const float p  = __expf(s - mn);
            l   = l * c + p;
            acc = acc * c + p * Vm[(size_t)j * DIM + h * HDIM + d];
            m = mn;
        }
    }
    outb[(size_t)i * DIM + h * HDIM + d] = bf_bits(acc / l);
}

// out = LayerNorm(a + b) * w + bb; optional bf16 copy for downstream GEMM.
__global__ void add_ln_kernel(const float* __restrict__ a,
                              const float* __restrict__ b,
                              const float* __restrict__ w,
                              const float* __restrict__ bb,
                              float* __restrict__ out,
                              unsigned short* __restrict__ outb)
{
    __shared__ float red[DIM];
    const int r = blockIdx.x, t = threadIdx.x;
    const float x = a[(size_t)r * DIM + t] + b[(size_t)r * DIM + t];

    red[t] = x; __syncthreads();
    for (int s = DIM / 2; s > 0; s >>= 1) {
        if (t < s) red[t] += red[t + s];
        __syncthreads();
    }
    const float mu = red[0] * (1.0f / DIM);
    __syncthreads();

    const float dv = x - mu;
    red[t] = dv * dv; __syncthreads();
    for (int s = DIM / 2; s > 0; s >>= 1) {
        if (t < s) red[t] += red[t + s];
        __syncthreads();
    }
    const float var = red[0] * (1.0f / DIM);
    const float y = dv * rsqrtf(var + 1e-5f) * w[t] + bb[t];
    out[(size_t)r * DIM + t] = y;
    if (outb) outb[(size_t)r * DIM + t] = bf_bits(y);
}

// ---------------------------------------------------------------------------
extern "C" void kernel_launch(void* const* d_in, const int* in_sizes, int n_in,
                              void* d_out, int out_size, void* d_ws, size_t ws_size,
                              hipStream_t stream)
{
    const float* node = (const float*)d_in[0];
    const float* ef   = (const float*)d_in[1];
    const int*   src  = (const int*)d_in[2];
    const int*   dst  = (const int*)d_in[3];
    const float* Wq = (const float*)d_in[4];  const float* bq = (const float*)d_in[5];
    const float* Wk = (const float*)d_in[6];  const float* bk = (const float*)d_in[7];
    const float* Wv = (const float*)d_in[8];  const float* bv = (const float*)d_in[9];
    const float* Wo = (const float*)d_in[10]; const float* bo = (const float*)d_in[11];
    const float* We = (const float*)d_in[12]; const float* be = (const float*)d_in[13];
    const float* n1w = (const float*)d_in[14]; const float* n1b = (const float*)d_in[15];
    const float* W1 = (const float*)d_in[16]; const float* b1 = (const float*)d_in[17];
    const float* W2 = (const float*)d_in[18]; const float* b2 = (const float*)d_in[19];
    const float* n2w = (const float*)d_in[20]; const float* n2b = (const float*)d_in[21];

    const size_t ND   = (size_t)NN_NODES * DIM;       // 524288
    const size_t NDFF = (size_t)NN_NODES * DFF;       // 2097152
    const size_t EH   = (size_t)NE * NH;              // 262144
    const size_t NNq  = (size_t)NN_NODES * NN_NODES;  // 4194304
    const size_t WDD  = (size_t)DIM * DIM;            // 65536
    const size_t WDF  = (size_t)DIM * DFF;            // 262144

    float* ws    = (float*)d_ws;
    float* Qf    = ws;                 // N x D fp32
    float* Kf    = Qf + ND;
    float* Vf    = Kf + ND;
    float* oproj = Vf + ND;            // N x D fp32
    float* x1    = oproj + ND;         // N x D fp32
    float* fout  = x1 + ND;            // N x D fp32
    float* eb    = fout + ND;          // E x H fp32
    int*   eid   = (int*)(eb + EH);    // N x N int
    unsigned char* adj = (unsigned char*)(eid + NNq);  // N x N bytes
    unsigned short* xb    = (unsigned short*)(adj + NNq);
    unsigned short* attnb = xb + ND;            // N x D bf16
    unsigned short* x1b   = attnb + ND;         // N x D bf16
    unsigned short* hbufb = x1b + ND;           // N x DFF bf16
    unsigned short* Wqp = hbufb + NDFF;         // packed bf16 weights
    unsigned short* Wkp = Wqp + WDD;
    unsigned short* Wvp = Wkp + WDD;
    unsigned short* Wop = Wvp + WDD;
    unsigned short* W1p = Wop + WDD;            // D x DFF
    unsigned short* W2p = W1p + WDF;            // DFF x D

    // ---- one-time conversions / weight packing (L2-resident, tiny) ----
    f32_to_bf16_kernel<<<(int)(ND / 256), 256, 0, stream>>>(node, xb, (int)ND);
    pack_b_kernel<<<(int)(WDD / 256), 256, 0, stream>>>(Wq, Wqp, DIM, DIM);
    pack_b_kernel<<<(int)(WDD / 256), 256, 0, stream>>>(Wk, Wkp, DIM, DIM);
    pack_b_kernel<<<(int)(WDD / 256), 256, 0, stream>>>(Wv, Wvp, DIM, DIM);
    pack_b_kernel<<<(int)(WDD / 256), 256, 0, stream>>>(Wo, Wop, DIM, DIM);
    pack_b_kernel<<<(int)(WDF / 256), 256, 0, stream>>>(W1, W1p, DIM, DFF);
    pack_b_kernel<<<(int)(WDF / 256), 256, 0, stream>>>(W2, W2p, DFF, DIM);

    const dim3 gblk(128);                       // 4 waves / block
    const int wavesD  = (NN_NODES / 16) * (DIM / 64);   // 512
    const int wavesF1 = (NN_NODES / 16) * (DFF / 64);   // 2048

    // ---- QKV projections (bf16 WMMA, fp32 out for attention math) ----
    wmma_gemm_kernel<DIM><<<wavesD / 4, gblk, 0, stream>>>(xb, Wqp, bq, Qf, nullptr, NN_NODES, DIM, 0);
    wmma_gemm_kernel<DIM><<<wavesD / 4, gblk, 0, stream>>>(xb, Wkp, bk, Kf, nullptr, NN_NODES, DIM, 0);
    wmma_gemm_kernel<DIM><<<wavesD / 4, gblk, 0, stream>>>(xb, Wvp, bv, Vf, nullptr, NN_NODES, DIM, 0);

    // ---- edge structure + bias ----
    init_graph_kernel<<<(int)(NNq / 256), 256, 0, stream>>>(eid, adj);
    scatter_graph_kernel<<<NE / 256, 256, 0, stream>>>(src, dst, eid, adj);
    edge_bias_kernel<<<(int)(EH / 256), 256, 0, stream>>>(ef, We, be, eb);

    // ---- sparse masked attention (online softmax) -> bf16 ----
    attn_kernel<<<NN_NODES, 256, 0, stream>>>(Qf, Kf, Vf, adj, eid, eb, attnb);

    // ---- output projection + residual + LN1 ----
    wmma_gemm_kernel<DIM><<<wavesD / 4, gblk, 0, stream>>>(attnb, Wop, bo, oproj, nullptr, NN_NODES, DIM, 0);
    add_ln_kernel<<<NN_NODES, DIM, 0, stream>>>(oproj, node, n1w, n1b, x1, x1b);

    // ---- FFN (WMMA) + residual + LN2 ----
    wmma_gemm_kernel<DIM><<<wavesF1 / 4, gblk, 0, stream>>>(x1b, W1p, b1, nullptr, hbufb, NN_NODES, DFF, 1);
    wmma_gemm_kernel<DFF><<<wavesD / 4, gblk, 0, stream>>>(hbufb, W2p, b2, fout, nullptr, NN_NODES, DIM, 0);
    add_ln_kernel<<<NN_NODES, DIM, 0, stream>>>(x1, fout, n2w, n2b, (float*)d_out, nullptr);
}
